// SpikingBlock_55576876810899
// MI455X (gfx1250) — matrix operand order repaired and checked by reading, not compiled, not run on hardware
//
#include <hip/hip_runtime.h>

#define T_DIM 2048
#define B_DIM 16
#define H_DIM 1024
#define BH    (B_DIM * H_DIM)        // 16384
#define M_DIM (T_DIM * B_DIM)        // 32768
#define CHUNK 256
#define NCHUNK (T_DIM / CHUNK)       // 8

typedef __attribute__((ext_vector_type(16))) _Float16 v16h;
typedef __attribute__((ext_vector_type(4)))  _Float16 v4h;
typedef __attribute__((ext_vector_type(8)))  float    v8f;
typedef __attribute__((ext_vector_type(4)))  float    f32x4;

union Frag16 { v16h h; f32x4 f[2]; };

// ---------------------------------------------------------------------------
// Pass 1: per-chunk local leaky scan. v_local[t] = beta*v_local[t-1] + x[t]
// (v_local[-1]=0 within each chunk). Store v_local as f16; store chunk-end
// value (f32) as carry for cross-chunk correction.
// ---------------------------------------------------------------------------
__global__ __launch_bounds__(256) void scan_pass1(const float* __restrict__ x,
                                                  const float* __restrict__ beta,
                                                  _Float16* __restrict__ vloc,
                                                  float* __restrict__ carry) {
    int tid   = blockIdx.x * 256 + threadIdx.x;   // over NCHUNK*BH
    int chunk = tid >> 14;                        // / 16384
    int idx   = tid & (BH - 1);
    int h     = idx & (H_DIM - 1);
    float bb  = beta[h];
    float v   = 0.0f;
    size_t base = (size_t)chunk * CHUNK * BH + (size_t)idx;
    const float* xp = x + base;
    _Float16*    vp = vloc + base;
#pragma unroll 4
    for (int tl = 0; tl < CHUNK; ++tl) {
        v = __builtin_fmaf(bb, v, *xp);
        *vp = (_Float16)v;
        xp += BH;
        vp += BH;
    }
    carry[chunk * BH + idx] = v;
}

// ---------------------------------------------------------------------------
// Pass 2: fold carries serially over the (8) chunks and apply the
// beta^(tl+1) * carry_in correction to each element of this chunk in place.
// ---------------------------------------------------------------------------
__global__ __launch_bounds__(256) void scan_pass2(const float* __restrict__ beta,
                                                  const float* __restrict__ carry,
                                                  _Float16* __restrict__ vloc) {
    int tid   = blockIdx.x * 256 + threadIdx.x;
    int chunk = tid >> 14;
    if (chunk == 0) return;                       // first chunk needs no carry
    int idx = tid & (BH - 1);
    int h   = idx & (H_DIM - 1);
    float bb = beta[h];
    // beta^CHUNK, CHUNK = 256 = 2^8 -> 8 squarings
    float bC = bb;
#pragma unroll
    for (int i = 0; i < 8; ++i) bC *= bC;
    // carry into this chunk: c_j = v_end_j + beta^CHUNK * c_{j-1}
    float c = 0.0f;
    for (int j = 0; j < chunk; ++j)
        c = __builtin_fmaf(bC, c, carry[j * BH + idx]);

    _Float16* vp = vloc + (size_t)chunk * CHUNK * BH + (size_t)idx;
    float s = bb;                                 // beta^(tl+1)
#pragma unroll 4
    for (int tl = 0; tl < CHUNK; ++tl) {
        float vv = (float)(*vp) + s * c;
        *vp = (_Float16)vv;
        s *= bb;
        vp += BH;
    }
}

// ---------------------------------------------------------------------------
// Convert W (f32, row-major [o][h]) to f16 once.
// ---------------------------------------------------------------------------
__global__ __launch_bounds__(256) void conv_w(const float* __restrict__ W,
                                              _Float16* __restrict__ Wh) {
    int i = blockIdx.x * 256 + threadIdx.x;       // over H*H/4
    f32x4 w = ((const f32x4*)W)[i];
    v4h o;
    o[0] = (_Float16)w[0];
    o[1] = (_Float16)w[1];
    o[2] = (_Float16)w[2];
    o[3] = (_Float16)w[3];
    ((v4h*)Wh)[i] = o;
}

// ---------------------------------------------------------------------------
// GEMM + bias + ReLU:  out[m][n] = relu(sum_k A[m][k] * W[n][k] + bias[n])
// A = v (f16, M_DIM x H_DIM), W f16 (H_DIM x H_DIM), out f32.
// Block = 256 threads = 8 wave32s arranged 2(M) x 4(N); block tile 128x256.
// Each wave: 4 M-subtiles x 4 N-subtiles of 16x16 (64x64), K stepped by 32:
// 16 b128 loads feed 16 WMMAs per K-step (1.0 loads/WMMA), 4x reuse of every
// A and B fragment. ~210 VGPRs, no LDS; operands stream from L2 (v = 64 MiB,
// W = 2 MiB both L2-resident on the 192 MB L2).
// ---------------------------------------------------------------------------
__global__ __launch_bounds__(256) void gemm_bias_relu(
        const _Float16* __restrict__ A,
        const _Float16* __restrict__ Bw,
        const float* __restrict__ bias,
        float* __restrict__ out) {
    const int lane  = threadIdx.x & 31;
    const int wave  = threadIdx.x >> 5;
    const int waveM = wave & 1;
    const int waveN = wave >> 1;
    const int rowBase = blockIdx.x * 128 + waveM * 64;
    const int colBase = blockIdx.y * 256 + waveN * 64;

    const int lm = lane & 15;   // M-row (A) / N-col (B,C,D) within tile
    const int lh = lane >> 4;   // half-wave selector

    v8f acc[4][4];
#pragma unroll
    for (int i = 0; i < 4; ++i)
#pragma unroll
        for (int j = 0; j < 4; ++j)
            acc[i][j] = (v8f){};

    // Per-lane base pointers.
    // A 16x32 f16 layout: lanes 0-15 hold K 0-7 (VGPR0-3) and 16-23 (VGPR4-7),
    // lanes 16-31 hold K 8-15 and 24-31, row M = lane%16.
    const _Float16* pa0 = A + (size_t)(rowBase + lm) * H_DIM + lh * 8;
    // B 32x16 f16 layout: lanes 0-15 hold K 0-7 and 16-23 for col N = lane,
    // lanes 16-31 hold K 8-15 and 24-31 for col N = lane-16 (same order as A).
    const _Float16* pb0 = Bw + (size_t)(colBase + lm) * H_DIM + lh * 8;

    for (int k0 = 0; k0 < H_DIM; k0 += 32) {
        Frag16 a[4], b[4];
#pragma unroll
        for (int i = 0; i < 4; ++i) {
            const _Float16* pa = pa0 + (size_t)(16 * i) * H_DIM + k0;
            a[i].f[0] = *(const f32x4*)(pa);
            a[i].f[1] = *(const f32x4*)(pa + 16);
        }
#pragma unroll
        for (int j = 0; j < 4; ++j) {
            const _Float16* pb = pb0 + (size_t)(16 * j) * H_DIM + k0;
            b[j].f[0] = *(const f32x4*)(pb);
            b[j].f[1] = *(const f32x4*)(pb + 16);
        }
#pragma unroll
        for (int i = 0; i < 4; ++i)
#pragma unroll
            for (int j = 0; j < 4; ++j)
                acc[i][j] = __builtin_amdgcn_wmma_f32_16x16x32_f16(
                    /*neg_a=*/false, a[i].h,
                    /*neg_b=*/false, b[j].h,
                    /*c_mod=*/(short)0, acc[i][j],
                    /*reuse_a=*/false, /*reuse_b=*/false);
    }

    // Epilogue. C/D 16x16 f32 layout: lane = N (both halves), VGPR r holds
    // M = r (lanes 0-15) or M = 8+r (lanes 16-31).
#pragma unroll
    for (int j = 0; j < 4; ++j) {
        const int col = colBase + 16 * j + lm;
        const float bv = bias[col];
#pragma unroll
        for (int i = 0; i < 4; ++i) {
            const int row0 = rowBase + 16 * i + lh * 8;
#pragma unroll
            for (int r = 0; r < 8; ++r) {
                float vv = acc[i][j][r] + bv;
                out[(size_t)(row0 + r) * H_DIM + col] = fmaxf(vv, 0.0f);
            }
        }
    }
}

// ---------------------------------------------------------------------------
extern "C" void kernel_launch(void* const* d_in, const int* in_sizes, int n_in,
                              void* d_out, int out_size, void* d_ws, size_t ws_size,
                              hipStream_t stream) {
    const float* x    = (const float*)d_in[0];   // (T, B, H) f32
    const float* beta = (const float*)d_in[1];   // (H,)      f32
    const float* W    = (const float*)d_in[2];   // (H, H)    f32
    const float* bias = (const float*)d_in[3];   // (H,)      f32
    float* out = (float*)d_out;                  // (T, B, H) f32

    // Workspace layout:
    //   vws   : M_DIM*H_DIM f16  (64 MiB)  — fits in 192 MB L2
    //   Wh    : H_DIM*H_DIM f16  ( 2 MiB)
    //   carry : NCHUNK*BH   f32  (512 KiB)
    _Float16* vws = (_Float16*)d_ws;
    _Float16* Wh  = vws + (size_t)M_DIM * H_DIM;
    float* carry  = (float*)(Wh + (size_t)H_DIM * H_DIM);

    conv_w<<<(H_DIM * H_DIM / 4) / 256, 256, 0, stream>>>(W, Wh);
    scan_pass1<<<(NCHUNK * BH) / 256, 256, 0, stream>>>(x, beta, vws, carry);
    scan_pass2<<<(NCHUNK * BH) / 256, 256, 0, stream>>>(beta, carry, vws);

    dim3 grid(M_DIM / 128, H_DIM / 256);
    gemm_bias_relu<<<grid, 256, 0, stream>>>(vws, Wh, bias, out);
}
